// Glm5NextLinearAttention_86784109183348
// MI455X (gfx1250) — matrix operand order repaired and checked by reading, not compiled, not run on hardware
//
#include <hip/hip_runtime.h>
#include <hip/hip_bf16.h>

#define T_TOK     4096
#define HIDDEN_D  2048
#define NUM_HEADS 32
#define HEAD_DIM  128
#define PROJ      (NUM_HEADS * HEAD_DIM)   // 4096
#define EPS_F     1e-5f
#define LOWER_BOUND_F -5.0f
#define BETA_SCALE_F   2.0f

typedef __attribute__((ext_vector_type(16))) __bf16 v16bf;
typedef __attribute__((ext_vector_type(8)))  __bf16 v8bf;
typedef __attribute__((ext_vector_type(8)))  float  v8f;
typedef __attribute__((ext_vector_type(4)))  unsigned int v4u;
typedef __attribute__((ext_vector_type(8)))  int v8i_;
typedef __attribute__((ext_vector_type(4)))  int v4i_;

__device__ __forceinline__ float sigmoidf_(float x) { return 1.0f / (1.0f + __expf(-x)); }
__device__ __forceinline__ float siluf_(float x)    { return x * sigmoidf_(x); }

// ---------------------------------------------------------------------------
// Tiled GEMM: C[M,N] = A[M,K] * B[K,N]; fp32 in memory, bf16 WMMA compute.
// Block tile 128x128, 8 waves; wave tile 64x32 = 4x2 WMMA accumulators.
//  - A tile: fetched by the Tensor Data Mover (tensor_load_to_lds) as raw
//    f32 with D#.pad_enable inserting 4 dwords per 32-dword row -> LDS row
//    stride 36 f32 (bank spread). Double buffered; wave 0 issues the DMA for
//    tile i+1 during compute of tile i; readiness via s_wait_tensorcnt(0).
//  - B tile: transposed [n][k] bf16 staging (coalesced strided global reads,
//    ds_store_b128), double buffered.
//  - ONE barrier per K-step (double buffering makes the 2nd barrier redundant).
// Requires M%128==0, N%128==0, K%32==0.
// ---------------------------------------------------------------------------
#define LDSS  40   // bf16 elements per B LDS row (32 data + 8 pad)
#define LDSA  36   // f32  elements per A LDS row (32 data + 4 pad via TDM)

__global__ __launch_bounds__(256) void wmma_gemm_tiled(
    const float* __restrict__ A, const float* __restrict__ B,
    float* __restrict__ C, int M, int N, int K)
{
    __shared__ __attribute__((aligned(16))) float  lAf[2][128 * LDSA];
    __shared__ __attribute__((aligned(16))) __bf16 lB [2][128 * LDSS];

    const int tid   = threadIdx.x;
    const int wave  = tid >> 5;
    const int lane  = tid & 31;
    const int lhalf = lane >> 4;
    const int l15   = lane & 15;

    const int nbn = N >> 7;
    const int bm  = blockIdx.x / nbn;
    const int bn  = blockIdx.x - bm * nbn;
    const int m0  = bm << 7;
    const int n0  = bn << 7;

    const int wm = (wave & 1) * 64;   // wave's M origin within block (4 tiles)
    const int wn = (wave >> 1) * 32;  // wave's N origin within block (2 tiles)

    // staging roles for B: 256 threads = 128 cols x 2 k-halves of 16
    const int sr = tid & 127;
    const int sk = (tid >> 7) * 16;

    // --- TDM descriptor issue for A tile [m0..m0+127, kk..kk+31] -> lAf[buf]
    auto issue_tdm_A = [&](int kk, int buf) {
        const unsigned lds = (unsigned)(size_t)&lAf[buf][0];
        const unsigned long long ga =
            (unsigned long long)(size_t)(A + (size_t)m0 * K + kk);
        v4u g0;
        g0[0] = 1u;                                        // count=1, user mode
        g0[1] = lds;                                       // lds_addr (bytes)
        g0[2] = (unsigned)(ga & 0xffffffffull);            // global_addr lo
        g0[3] = (unsigned)((ga >> 32) & 0x01ffffffull)     // global_addr hi
              | (2u << 30);                                // type = 2 (image)
        v8i_ g1;
        g1[0] = (int)((2u << 16)      // data_size = 4 bytes
                    | (1u << 20)      // pad_enable
                    | (4u << 22)      // pad_interval: every 32 dwords
                    | (3u << 25));    // pad_amount: 4 dwords
        g1[1] = (int)(((unsigned)K & 0xffffu) << 16);                  // td0 lo
        g1[2] = (int)((((unsigned)K >> 16) & 0xffffu)
                    | (((unsigned)M & 0xffffu) << 16));                // td0 hi | td1 lo
        g1[3] = (int)((((unsigned)M >> 16) & 0xffffu) | (32u << 16));  // td1 hi | tile_dim0=32
        g1[4] = (int)128u;                                 // tile_dim1=128, tile_dim2=0
        g1[5] = (int)K;                                    // tensor_dim0_stride lo (elems)
        g1[6] = 0;
        g1[7] = 0;
        v4i_ g2 = {};
        v4i_ g3 = {};
        v8i_ g4 = {};                                      // unused trailing group
        __builtin_amdgcn_tensor_load_to_lds(g0, g1, g2, g3, g4, 0);
    };

    v8f acc[4][2];
#pragma unroll
    for (int mt = 0; mt < 4; ++mt)
#pragma unroll
        for (int nt = 0; nt < 2; ++nt)
            acc[mt][nt] = (v8f){};

    // prologue: kick off DMA of the first A tile
    if (wave == 0) issue_tdm_A(0, 0);

    for (int kk = 0; kk < K; kk += 32) {
        const int buf = (kk >> 5) & 1;

        // ---- stage B tile transposed: col n0+sr, K kk+sk..+15 (coalesced)
        {
            const float* gb = B + (size_t)(kk + sk) * N + n0 + sr;
            if (kk + 32 < K) __builtin_prefetch(gb + (size_t)32 * N, 0, 0);
            float f[16];
#pragma unroll
            for (int j = 0; j < 16; ++j) f[j] = gb[(size_t)j * N];
            v8bf lo, hi;
#pragma unroll
            for (int j = 0; j < 8; ++j) { lo[j] = (__bf16)f[j]; hi[j] = (__bf16)f[j+8]; }
            *(v8bf*)&lB[buf][sr * LDSS + sk]     = lo;
            *(v8bf*)&lB[buf][sr * LDSS + sk + 8] = hi;
        }

        // A tile DMA for this step must have landed
        if (wave == 0) __builtin_amdgcn_s_wait_tensorcnt((short)0);
        __syncthreads();

        // overlap: DMA the next A tile while this one is consumed
        if (wave == 0 && kk + 32 < K) issue_tdm_A(kk + 32, buf ^ 1);

        // ---- A fragments: f32 from LDS, convert to bf16 at load time
        v16bf af[4], bfr[2];
        const int koff = lhalf * 8;                 // K {0..7} vs {8..15}
#pragma unroll
        for (int mt = 0; mt < 4; ++mt) {
            const int row = wm + mt * 16 + l15;
            const float* ap = &lAf[buf][row * LDSA + koff];
            const float4 x0 = ((const float4*)ap)[0];
            const float4 x1 = ((const float4*)ap)[1];
            const float4 x2 = ((const float4*)(ap + 16))[0];
            const float4 x3 = ((const float4*)(ap + 16))[1];
            v16bf a;
            a[0]=(__bf16)x0.x; a[1]=(__bf16)x0.y; a[2]=(__bf16)x0.z; a[3]=(__bf16)x0.w;
            a[4]=(__bf16)x1.x; a[5]=(__bf16)x1.y; a[6]=(__bf16)x1.z; a[7]=(__bf16)x1.w;
            a[8]=(__bf16)x2.x; a[9]=(__bf16)x2.y; a[10]=(__bf16)x2.z; a[11]=(__bf16)x2.w;
            a[12]=(__bf16)x3.x; a[13]=(__bf16)x3.y; a[14]=(__bf16)x3.z; a[15]=(__bf16)x3.w;
            af[mt] = a;
        }
#pragma unroll
        for (int nt = 0; nt < 2; ++nt) {
            const int n  = wn + nt * 16 + l15;
            const int kb = lhalf * 8;               // K {0..7,16..23} vs {8..15,24..31}
            const v8bf c1 = *(const v8bf*)&lB[buf][n * LDSS + kb];
            const v8bf c2 = *(const v8bf*)&lB[buf][n * LDSS + kb + 16];
            bfr[nt] = __builtin_shufflevector(c1, c2, 0,1,2,3,4,5,6,7,
                                                      8,9,10,11,12,13,14,15);
        }

        // ---- 8 WMMAs with register reuse
#pragma unroll
        for (int mt = 0; mt < 4; ++mt)
#pragma unroll
            for (int nt = 0; nt < 2; ++nt)
                acc[mt][nt] = __builtin_amdgcn_wmma_f32_16x16x32_bf16(
                    false, af[mt], false, bfr[nt], (short)0, acc[mt][nt],
                    false, false);
        // no trailing barrier: next iteration writes the other buffer, and
        // its barrier orders reuse two steps out.
    }

    // ---- epilogue: VGPR r -> row (+lhalf*8 + r), col l15
#pragma unroll
    for (int mt = 0; mt < 4; ++mt)
#pragma unroll
        for (int nt = 0; nt < 2; ++nt) {
            float* Cp = C + (size_t)(m0 + wm + mt * 16 + lhalf * 8) * N
                          + n0 + wn + nt * 16 + l15;
#pragma unroll
            for (int r = 0; r < 8; ++r)
                Cp[(size_t)r * N] = acc[mt][nt][r];
        }
}

// ---------------------------------------------------------------------------
// Single-tile WMMA GEMM for small N (beta projection, N=32).
// ---------------------------------------------------------------------------
__global__ __launch_bounds__(256) void wmma_gemm_small(
    const float* __restrict__ A, const float* __restrict__ B,
    float* __restrict__ C, int M, int N, int K)
{
    const int wave = threadIdx.x >> 5;
    const int lane = threadIdx.x & 31;
    const int ntn  = N >> 4;
    const int tile = blockIdx.x * 8 + wave;
    const int tm   = tile / ntn;
    const int tn   = tile - tm * ntn;
    const int m0   = tm << 4, n0 = tn << 4;
    const int lhalf = lane >> 4, l15 = lane & 15;

    const float* Arow = A + (size_t)(m0 + l15) * K + (lhalf ? 16 : 0);
    const float* Bcol = B + (size_t)n0 + l15;
    const int bko = lhalf ? 16 : 0;

    v8f acc = {};
    for (int kk = 0; kk < K; kk += 32) {
        v16bf a, b;
#pragma unroll
        for (int j = 0; j < 8; ++j) {
            a[j]     = (__bf16)Arow[kk + j];
            a[j + 8] = (__bf16)Arow[kk + 8 + j];
        }
#pragma unroll
        for (int j = 0; j < 16; ++j)
            b[j] = (__bf16)Bcol[(size_t)(kk + bko + j) * N];
        acc = __builtin_amdgcn_wmma_f32_16x16x32_bf16(
            false, a, false, b, (short)0, acc, false, false);
    }
    float* Crow = C + (size_t)(m0 + (lhalf ? 8 : 0)) * N + n0 + l15;
#pragma unroll
    for (int r = 0; r < 8; ++r) Crow[(size_t)r * N] = acc[r];
}

// ---------------------------------------------------------------------------
// Depthwise causal conv (K=4) + SiLU (+ optional scale). x,y:[T,PROJ]; w:[PROJ,4]
// ---------------------------------------------------------------------------
__global__ __launch_bounds__(256) void conv_silu_k(
    const float* __restrict__ x, const float* __restrict__ w,
    float* __restrict__ y, float outScale)
{
    const int idx = blockIdx.x * 256 + threadIdx.x;
    const int t   = idx >> 12;
    const int p   = idx & (PROJ - 1);
    float acc = 0.0f;
#pragma unroll
    for (int j = 0; j < 4; ++j) {
        const int tt = t - 3 + j;
        const float xv = (tt >= 0) ? x[(size_t)tt * PROJ + p] : 0.0f;
        acc += xv * w[p * 4 + j];
    }
    y[idx] = siluf_(acc) * outScale;
}

// g1 = LOWER_BOUND * sigmoid(exp(A_log[h]) * (g1pre + dt_bias[p]))  (in place)
__global__ __launch_bounds__(256) void g1_gate_k(
    float* __restrict__ g1, const float* __restrict__ dt_bias,
    const float* __restrict__ A_log)
{
    const int idx = blockIdx.x * 256 + threadIdx.x;
    const int p = idx & (PROJ - 1);
    const int h = p >> 7;
    const float v = g1[idx] + dt_bias[p];
    g1[idx] = LOWER_BOUND_F * sigmoidf_(__expf(A_log[h]) * v);
}

__global__ __launch_bounds__(256) void beta_gate_k(float* __restrict__ b)
{
    const int idx = blockIdx.x * 256 + threadIdx.x;
    b[idx] = sigmoidf_(b[idx]) * BETA_SCALE_F;
}

// ---------------------------------------------------------------------------
// KDA gated delta-rule scan; independent per value column.
// grid = 32 heads x 4 column groups; 256 threads = 32 cols x 8 k-segments.
// ---------------------------------------------------------------------------
__global__ __launch_bounds__(256) void kda_scan_k(
    const float* __restrict__ q, const float* __restrict__ k,
    const float* __restrict__ v, const float* __restrict__ g,
    const float* __restrict__ beta, float* __restrict__ o)
{
    const int h    = blockIdx.x >> 2;
    const int cg   = blockIdx.x & 3;
    const int tid  = threadIdx.x;
    const int lc   = tid & 31;
    const int col  = cg * 32 + lc;
    const int ksub = tid >> 5;
    const int kr0  = ksub * 16;

    __shared__ float sh_k[HEAD_DIM];
    __shared__ float sh_q[HEAD_DIM];
    __shared__ float sh_eg[HEAD_DIM];
    __shared__ float sh_u[32];
    __shared__ float sh_red[256];
    __shared__ float sh_beta;

    float S[16];
#pragma unroll
    for (int i = 0; i < 16; ++i) S[i] = 0.0f;

    for (int t = 0; t < T_TOK; ++t) {
        const size_t base = ((size_t)t * NUM_HEADS + h) * HEAD_DIM;
        if (tid < HEAD_DIM) {
            sh_k[tid]  = k[base + tid];
            sh_q[tid]  = q[base + tid];
            sh_eg[tid] = __expf(g[base + tid]);
        }
        if (tid == 128) sh_beta = beta[(size_t)t * NUM_HEADS + h];
        __syncthreads();

        float ks = 0.0f;
#pragma unroll
        for (int i = 0; i < 16; ++i) {
            const float s = S[i] * sh_eg[kr0 + i];
            S[i] = s;
            ks += sh_k[kr0 + i] * s;
        }
        sh_red[tid] = ks;
        __syncthreads();

        if (ksub == 0) {
            float tot = 0.0f;
#pragma unroll
            for (int j = 0; j < 8; ++j) tot += sh_red[lc + 32 * j];
            sh_u[lc] = sh_beta * (v[base + col] - tot);
        }
        __syncthreads();

        const float u = sh_u[lc];
        float oa = 0.0f;
#pragma unroll
        for (int i = 0; i < 16; ++i) {
            const float s = S[i] + sh_k[kr0 + i] * u;
            S[i] = s;
            oa += sh_q[kr0 + i] * s;
        }
        sh_red[tid] = oa;
        __syncthreads();

        if (ksub == 0) {
            float tot = 0.0f;
#pragma unroll
            for (int j = 0; j < 8; ++j) tot += sh_red[lc + 32 * j];
            o[base + col] = tot;
        }
        __syncthreads();
    }
}

// ---------------------------------------------------------------------------
// Gated RMSNorm: one wave per (t,h) row; 4 elems/lane, shfl_xor reduction.
// ---------------------------------------------------------------------------
__global__ __launch_bounds__(256) void rmsnorm_gate_k(
    float* __restrict__ o, const float* __restrict__ g2,
    const float* __restrict__ w)
{
    const int row  = blockIdx.x * 8 + (threadIdx.x >> 5);
    const int lane = threadIdx.x & 31;
    const size_t base = (size_t)row * HEAD_DIM + lane * 4;

    float4 ov = *(const float4*)(o + base);
    float ss = ov.x * ov.x + ov.y * ov.y + ov.z * ov.z + ov.w * ov.w;
#pragma unroll
    for (int off = 16; off > 0; off >>= 1) ss += __shfl_xor(ss, off, 32);

    const float r = rsqrtf(ss * (1.0f / HEAD_DIM) + EPS_F);
    const float4 gv = *(const float4*)(g2 + base);
    const float4 wv = *(const float4*)(w + lane * 4);

    float4 out;
    out.x = ov.x * r * wv.x * sigmoidf_(gv.x);
    out.y = ov.y * r * wv.y * sigmoidf_(gv.y);
    out.z = ov.z * r * wv.z * sigmoidf_(gv.z);
    out.w = ov.w * r * wv.w * sigmoidf_(gv.w);
    *(float4*)(o + base) = out;
}

// ---------------------------------------------------------------------------
extern "C" void kernel_launch(void* const* d_in, const int* in_sizes, int n_in,
                              void* d_out, int out_size, void* d_ws, size_t ws_size,
                              hipStream_t stream)
{
    const float* hs   = (const float*)d_in[0];
    const float* Wq   = (const float*)d_in[1];
    const float* Wk   = (const float*)d_in[2];
    const float* Wv   = (const float*)d_in[3];
    const float* cq   = (const float*)d_in[4];
    const float* ck   = (const float*)d_in[5];
    const float* cv   = (const float*)d_in[6];
    const float* Wb   = (const float*)d_in[7];
    const float* Wfa  = (const float*)d_in[8];
    const float* Wfb  = (const float*)d_in[9];
    const float* dtb  = (const float*)d_in[10];
    const float* Alog = (const float*)d_in[11];
    const float* Wga  = (const float*)d_in[12];
    const float* Wgb  = (const float*)d_in[13];
    const float* onw  = (const float*)d_in[14];
    const float* Wo   = (const float*)d_in[15];

    char* ws = (char*)d_ws;
    const size_t SZ_TP = (size_t)T_TOK * PROJ * sizeof(float);        // 64 MB
    float* bufA    = (float*)(ws + 0 * SZ_TP);
    float* bufQ    = (float*)(ws + 1 * SZ_TP);
    float* bufK    = (float*)(ws + 2 * SZ_TP);
    float* bufV    = (float*)(ws + 3 * SZ_TP);
    float* bufG2   = (float*)(ws + 4 * SZ_TP);
    float* bufO    = (float*)(ws + 5 * SZ_TP);
    float* bufFa   = (float*)(ws + 6 * SZ_TP);
    float* bufBeta = (float*)(ws + 6 * SZ_TP +
                              (size_t)T_TOK * HEAD_DIM * sizeof(float));

    auto gemm = [&](const float* A, const float* B, float* C, int M, int N, int K) {
        const int blocks = (M >> 7) * (N >> 7);
        wmma_gemm_tiled<<<dim3(blocks), dim3(256), 0, stream>>>(A, B, C, M, N, K);
    };

    const float qscale = 0.08838834764831845f;   // HEAD_DIM^-0.5
    const int ew_grid = T_TOK * PROJ / 256;

    // q / k / v projections + causal conv + silu
    gemm(hs, Wq, bufA, T_TOK, PROJ, HIDDEN_D);
    conv_silu_k<<<ew_grid, 256, 0, stream>>>(bufA, cq, bufQ, qscale);
    gemm(hs, Wk, bufA, T_TOK, PROJ, HIDDEN_D);
    conv_silu_k<<<ew_grid, 256, 0, stream>>>(bufA, ck, bufK, 1.0f);
    gemm(hs, Wv, bufA, T_TOK, PROJ, HIDDEN_D);
    conv_silu_k<<<ew_grid, 256, 0, stream>>>(bufA, cv, bufV, 1.0f);

    // g1 low-rank gate
    gemm(hs, Wfa, bufFa, T_TOK, HEAD_DIM, HIDDEN_D);
    gemm(bufFa, Wfb, bufA, T_TOK, PROJ, HEAD_DIM);
    g1_gate_k<<<ew_grid, 256, 0, stream>>>(bufA, dtb, Alog);

    // g2 output gate
    gemm(hs, Wga, bufFa, T_TOK, HEAD_DIM, HIDDEN_D);
    gemm(bufFa, Wgb, bufG2, T_TOK, PROJ, HEAD_DIM);

    // beta (N=32 -> small-tile kernel)
    {
        const int tiles = (T_TOK / 16) * (NUM_HEADS / 16);
        wmma_gemm_small<<<dim3(tiles / 8), 256, 0, stream>>>(
            hs, Wb, bufBeta, T_TOK, NUM_HEADS, HIDDEN_D);
    }
    beta_gate_k<<<T_TOK * NUM_HEADS / 256, 256, 0, stream>>>(bufBeta);

    // sequential gated delta-rule scan: 32 heads x 4 column groups
    kda_scan_k<<<NUM_HEADS * 4, 256, 0, stream>>>(bufQ, bufK, bufV, bufA,
                                                  bufBeta, bufO);

    // gated RMSNorm (in place)
    rmsnorm_gate_k<<<T_TOK * NUM_HEADS / 8, 256, 0, stream>>>(bufO, bufG2, onw);

    // output projection -> d_out
    gemm(bufO, Wo, (float*)d_out, T_TOK, HIDDEN_D, PROJ);
}
